// GraphConvLayer_18330920419716
// MI455X (gfx1250) — matrix operand order repaired and checked by reading, not compiled, not run on hardware
//
#include <hip/hip_runtime.h>
#include <hip/hip_bf16.h>
#include <stdint.h>

// ---------------------------------------------------------------------------
// GraphConv: out = segment_sum( (X@W)[col] * val, row ) + bias
//   Kernel 0: Wt[n][k] = bf16(W[k][n])          (one-time 128KB transpose)
//   Kernel 1: support = X @ W via v_wmma_f32_16x16x32_bf16; whole Wt resident
//             in LDS (async-to-LDS preload), A frags loaded straight from
//             global in WMMA lane layout -> barrier-free inner loop.
//   Kernel 2: out[r][:] = bias[:]
//   Kernel 3: edge-parallel scatter with global_atomic_add_f32.
// ---------------------------------------------------------------------------

typedef __attribute__((ext_vector_type(16))) __bf16 v16bf;
typedef __attribute__((ext_vector_type(8)))  __bf16 v8bf;
typedef __attribute__((ext_vector_type(8)))  float  v8f;

#define KDIM 256          // inner dim (IN_DIM)
#define NDIM 256          // output dim (OUT_DIM)
#define TM   64           // M rows per block
#define KC   32           // K chunk per WMMA (bf16 K)
#define LDW  264          // padded LDS row stride (bf16): 528B = 33*16B ->
                          // 16B-aligned b128 frag reads, 16-bank spread

// Async global->LDS copy of 16B per lane (ASYNCcnt-tracked DMA path).
__device__ __forceinline__ void async_copy_b128(unsigned lds_off, const void* gptr) {
  asm volatile("global_load_async_to_lds_b128 %0, %1, off"
               :: "v"(lds_off), "v"((unsigned long long)(uintptr_t)gptr)
               : "memory");
}
__device__ __forceinline__ void wait_async0() {
  asm volatile("s_wait_asynccnt 0" ::: "memory");
}

// ---------------------------------------------------------------------------
// Kernel 0: Wt[n][k] = bf16(W[k][n])   (256x256, tight 512B rows)
// ---------------------------------------------------------------------------
__global__ __launch_bounds__(256) void gcn_wconv(
    const float* __restrict__ W, __bf16* __restrict__ Wt)
{
  const int t = blockIdx.x * 256 + threadIdx.x;   // 65536 elements
  const int k = t & 255;
  const int n = t >> 8;
  Wt[(size_t)n * 256 + k] = (__bf16)W[(size_t)k * 256 + n];
}

// ---------------------------------------------------------------------------
// Kernel 1: support = X @ W   (bf16 WMMA, f32 accumulate)
// Block: 256 threads = 8 waves; block tile 64(M) x 256(N).
// Wave w owns 2 m-tiles x 4 n-tiles; whole Wt lives in LDS; A frags loaded
// per-lane from global directly in WMMA layout -> no barriers in the k-loop.
// ---------------------------------------------------------------------------
__global__ __launch_bounds__(256) void gcn_gemm_wmma(
    const float* __restrict__ X, const __bf16* __restrict__ Wt,
    float* __restrict__ S, int M)
{
  __shared__ __align__(16) __bf16 Bsm[NDIM][LDW];   // 132 KB, [n][k] padded

  const int tid  = threadIdx.x;
  const int lane = tid & 31;
  const int wave = tid >> 5;
  const int m0b  = blockIdx.x * TM;

  const int mt0    = (wave & 1) * 2;    // first of two m-tiles (0 or 2)
  const int ntBase = (wave >> 1) * 4;   // first of four n-tiles (0,4,8,12)
  const int lhalf  = (lane >> 4);       // 0: lanes 0-15, 1: lanes 16-31
  const int l15    = lane & 15;

  // ---- preload entire Wt into LDS via async DMA: 1 row (512B) per wave/pass
  #pragma unroll 4
  for (int pass = 0; pass < 32; ++pass) {
    const int row = pass * 8 + wave;
    const void* gsrc = Wt + (size_t)row * NDIM + lane * 8;          // 16B/lane
    const unsigned ldst = (unsigned)(uintptr_t)&Bsm[row][lane * 8]; // LDS offset
    async_copy_b128(ldst, gsrc);
  }
  wait_async0();
  __syncthreads();

  // ---- A rows for this wave's two m-tiles (clamped; masked at store) ----
  int rowM[2];
  #pragma unroll
  for (int mi = 0; mi < 2; ++mi) {
    int r = m0b + (mt0 + mi) * 16 + l15;
    rowM[mi] = (r < M) ? r : (M - 1);
  }
  const int abase  = lhalf * 8;    // A frag K runs: abase..+7 and abase+16..+23
  const int bkhalf = lhalf * 8;    // B frag K runs: bkhalf..+7 and bkhalf+16..+23
  int bnRow[4];
  #pragma unroll
  for (int nt = 0; nt < 4; ++nt) bnRow[nt] = (ntBase + nt) * 16 + l15;

  v8f acc[2][4] = {};

  for (int kc = 0; kc < KDIM; kc += KC) {
    // ---- A fragments: per-lane global loads in WMMA layout + packed cvt ----
    union { v16bf v; v8bf h[2]; } au[2];
    #pragma unroll
    for (int mi = 0; mi < 2; ++mi) {
      const float* p = X + (size_t)rowM[mi] * KDIM + kc + abase;
      if (mi == 0 && kc + KC < KDIM)
        __builtin_prefetch(p + KC, 0, 3);
      const float4 f0 = *(const float4*)(p);        // K abase..abase+3
      const float4 f1 = *(const float4*)(p + 4);    // K abase+4..abase+7
      const float4 g0 = *(const float4*)(p + 16);   // K abase+16..abase+19
      const float4 g1 = *(const float4*)(p + 20);   // K abase+20..abase+23
      v8bf lo, hi;
      lo[0] = (__bf16)f0.x; lo[1] = (__bf16)f0.y;
      lo[2] = (__bf16)f0.z; lo[3] = (__bf16)f0.w;
      lo[4] = (__bf16)f1.x; lo[5] = (__bf16)f1.y;
      lo[6] = (__bf16)f1.z; lo[7] = (__bf16)f1.w;
      hi[0] = (__bf16)g0.x; hi[1] = (__bf16)g0.y;
      hi[2] = (__bf16)g0.z; hi[3] = (__bf16)g0.w;
      hi[4] = (__bf16)g1.x; hi[5] = (__bf16)g1.y;
      hi[6] = (__bf16)g1.z; hi[7] = (__bf16)g1.w;
      au[mi].h[0] = lo;
      au[mi].h[1] = hi;
    }

    // ---- 4 n-tiles: B frags from resident LDS, 2 WMMAs per frag ----
    #pragma unroll
    for (int nt = 0; nt < 4; ++nt) {
      const v8bf* bp = (const v8bf*)&Bsm[bnRow[nt]][kc + bkhalf];
      union { v16bf v; v8bf h[2]; } bu;
      bu.h[0] = bp[0];
      bu.h[1] = bp[2];
      acc[0][nt] = __builtin_amdgcn_wmma_f32_16x16x32_bf16(
          false, au[0].v, false, bu.v, (short)0, acc[0][nt], false, false);
      acc[1][nt] = __builtin_amdgcn_wmma_f32_16x16x32_bf16(
          false, au[1].v, false, bu.v, (short)0, acc[1][nt], false, false);
    }
  }

  // ---- store D: lanes 0-15 -> rows +0..7, lanes 16-31 -> rows +8..15 ----
  const int mrowBase = m0b + mt0 * 16 + lhalf * 8;
  if (m0b + TM <= M) {
    // Fast path: one base pointer, 64 stores with compile-time offsets.
    float* base = S + (size_t)mrowBase * NDIM + ntBase * 16 + l15;
    #pragma unroll
    for (int mi = 0; mi < 2; ++mi)
      #pragma unroll
      for (int nt = 0; nt < 4; ++nt)
        #pragma unroll
        for (int i = 0; i < 8; ++i)
          base[mi * 16 * NDIM + i * NDIM + nt * 16] = acc[mi][nt][i];
  } else {
    #pragma unroll
    for (int mi = 0; mi < 2; ++mi) {
      const int mrow = mrowBase + mi * 16;
      #pragma unroll
      for (int nt = 0; nt < 4; ++nt) {
        const int col = (ntBase + nt) * 16 + l15;
        #pragma unroll
        for (int i = 0; i < 8; ++i) {
          const int r = mrow + i;
          if (r < M) S[(size_t)r * NDIM + col] = acc[mi][nt][i];
        }
      }
    }
  }
}

// ---------------------------------------------------------------------------
// Kernel 2: out[r][c] = bias[c]   (float4 stores)
// ---------------------------------------------------------------------------
__global__ __launch_bounds__(256) void gcn_bias_init(
    float* __restrict__ out, const float* __restrict__ bias, int M)
{
  const int t = blockIdx.x * 256 + threadIdx.x;   // one float4 per thread
  if (t >= M * 64) return;
  const int r = t >> 6;
  const int c = (t & 63) * 4;
  const float4 b = *(const float4*)(bias + c);
  *(float4*)(out + (size_t)r * NDIM + c) = b;
}

// ---------------------------------------------------------------------------
// Kernel 3: one wave per edge; lane handles 8 dims (2x float4 load + 8 atomics)
// ---------------------------------------------------------------------------
__global__ __launch_bounds__(256) void gcn_spmm_scatter(
    const float* __restrict__ S, float* __restrict__ out,
    const int* __restrict__ row_idx, const int* __restrict__ col_idx,
    const float* __restrict__ vals, int E)
{
  const int e = blockIdx.x * 8 + (threadIdx.x >> 5);
  if (e >= E) return;
  const int lane = threadIdx.x & 31;

  const int   r = row_idx[e];
  const int   c = col_idx[e];
  const float v = vals[e];

  const float4* src = (const float4*)(S + (size_t)c * NDIM) + lane * 2;
  const float4 a = src[0];
  const float4 b = src[1];
  float* dst = out + (size_t)r * NDIM + lane * 8;

  atomicAdd(dst + 0, a.x * v);
  atomicAdd(dst + 1, a.y * v);
  atomicAdd(dst + 2, a.z * v);
  atomicAdd(dst + 3, a.w * v);
  atomicAdd(dst + 4, b.x * v);
  atomicAdd(dst + 5, b.y * v);
  atomicAdd(dst + 6, b.z * v);
  atomicAdd(dst + 7, b.w * v);
}

// ---------------------------------------------------------------------------
extern "C" void kernel_launch(void* const* d_in, const int* in_sizes, int n_in,
                              void* d_out, int out_size, void* d_ws, size_t ws_size,
                              hipStream_t stream) {
  const float* X    = (const float*)d_in[0];  // [M,256]
  const float* W    = (const float*)d_in[1];  // [256,256]
  const float* bias = (const float*)d_in[2];  // [256]
  const int* row    = (const int*)d_in[3];    // [E]
  const int* col    = (const int*)d_in[4];    // [E]
  const float* vals = (const float*)d_in[5];  // [E]

  const int M = in_sizes[0] / KDIM;
  const int E = in_sizes[3];

  float* out = (float*)d_out;
  float*  S  = (float*)d_ws;                          // support, M*256 f32
  __bf16* Wt = (__bf16*)((char*)d_ws + (size_t)M * NDIM * sizeof(float));

  // 0) Wt = bf16(W^T)
  gcn_wconv<<<256, 256, 0, stream>>>(W, Wt);
  // 1) support = X @ W  (bf16 WMMA, Wt resident in LDS)
  gcn_gemm_wmma<<<(M + TM - 1) / TM, 256, 0, stream>>>(X, Wt, S, M);
  // 2) out = bias (broadcast)
  gcn_bias_init<<<(M * 64 + 255) / 256, 256, 0, stream>>>(out, bias, M);
  // 3) out[row] += vals * support[col]
  gcn_spmm_scatter<<<(E + 7) / 8, 256, 0, stream>>>(S, out, row, col, vals, E);
}
